// SeqAttentionPool_57303453663827
// MI455X (gfx1250) — matrix operand order repaired and checked by reading, not compiled, not run on hardware
//
#include <hip/hip_runtime.h>
#include <type_traits>

// ---------------------------------------------------------------------------
// MHA forward for MI455X (gfx1250, wave32, WMMA f32_16x16x32_f16).
//   B=8, LQ=16, LK=8192, HIDDEN=1024, 16 heads x 64 dim.
// Pipeline: cvt weights -> Q/K/V projections (WMMA GEMM, f16 out, V stored
// transposed) -> fused flash attention (1 block per (b,h)) -> O projection.
// ---------------------------------------------------------------------------

typedef __attribute__((ext_vector_type(16))) _Float16 v16h;
typedef __attribute__((ext_vector_type(8)))  _Float16 v8h;
typedef __attribute__((ext_vector_type(8)))  float    v8f;

#define HIDDEN   1024
#define HEADS    16
#define HEAD_DIM 64
#define BATCH    8
#define LQ       16
#define LK       8192
#define BLK      (BATCH * LK)   // 65536 total key rows
#define MQ       (BATCH * LQ)   // 128 total query rows

__device__ __forceinline__ v16h cat8(v8h lo, v8h hi) {
  return __builtin_shufflevector(lo, hi, 0,1,2,3,4,5,6,7,8,9,10,11,12,13,14,15);
}

// A operand fragment (16x32 f16, M x K).  p = this lane's row at k-offset 0.
// Layout (ISA 7.12.2): lanes 0-15 hold K {0..7, 16..23}; lanes 16-31 hold
// K {8..15, 24..31}.
__device__ __forceinline__ v16h frag_a(const _Float16* p, int lane) {
  const int kb = (lane >> 4) * 8;
  v8h lo = *(const v8h*)(p + kb);
  v8h hi = *(const v8h*)(p + kb + 16);
  return cat8(lo, hi);
}

// B operand fragment (32x16 f16, K x N), source stored as N rows of
// K-contiguous f16.  Lanes 0-15: K {0..7, 16..23} of column l15; lanes 16-31: K {8..15, 24..31}.
__device__ __forceinline__ v16h frag_b(const _Float16* p, int lane) {
  const int kb = (lane >> 4) * 8;
  v8h lo = *(const v8h*)(p + kb);
  v8h hi = *(const v8h*)(p + kb + 16);
  return cat8(lo, hi);
}

// ---------------------------------------------------------------------------
__global__ void cvt_f32_f16(const float* __restrict__ in,
                            _Float16* __restrict__ out, int n) {
  for (int i = blockIdx.x * blockDim.x + threadIdx.x; i < n;
       i += gridDim.x * blockDim.x)
    out[i] = (_Float16)in[i];
}

// ---------------------------------------------------------------------------
// Out[m][n] = (sum_k X[m][k] * W16[n][k] + bias[n]) * scale
// Block tile: 128(M) x 64(N), BK=32.  8 waves: wave (wm,wn) owns a 32x32
// patch = 2x2 WMMA tiles.  M,N,K assumed multiples of 128/64/32.
// TRANS_OUT stores f16 Out^T (pitch ldo) with 16-byte packed stores.
// ---------------------------------------------------------------------------
template <typename XT, bool OUT_F32, bool TRANS_OUT>
__global__ __launch_bounds__(256)
void gemm_wmma(const XT* __restrict__ X, const _Float16* __restrict__ W16,
               const float* __restrict__ bias, void* __restrict__ OutP,
               int M, int N, int K, int ldo, float scale) {
  (void)M;
  __shared__ alignas(16) _Float16 sX[128][40];   // +8 pad: 20-bank row pitch
  __shared__ alignas(16) _Float16 sW[64][40];

  const int tid  = threadIdx.x;
  const int lane = tid & 31;
  const int w    = tid >> 5;
  const int l15  = lane & 15;
  const int half = lane >> 4;
  const int wm   = w & 3;                 // 4 M quadrants of 32 rows
  const int wn   = w >> 2;                // 2 N halves of 32 cols
  const int bm0  = blockIdx.x * 128;
  const int bn0  = blockIdx.y * 64;

  v8f acc[2][2] = {};

  const int xrow = tid >> 1;              // 0..127
  const int xks  = (tid & 1) * 16;
  const int wrow = tid >> 2;              // 0..63
  const int wks  = (tid & 3) * 8;

  for (int k0 = 0; k0 < K; k0 += 32) {
    // ---- stage X tile (128 x 32), converting fp32 -> f16 on the fly ----
    const XT* xp = X + (size_t)(bm0 + xrow) * K + k0 + xks;
    if constexpr (std::is_same<XT, float>::value) {
      float4 x0 = ((const float4*)xp)[0];
      float4 x1 = ((const float4*)xp)[1];
      float4 x2 = ((const float4*)xp)[2];
      float4 x3 = ((const float4*)xp)[3];
      v8h h0, h1;
      h0[0]=(_Float16)x0.x; h0[1]=(_Float16)x0.y; h0[2]=(_Float16)x0.z; h0[3]=(_Float16)x0.w;
      h0[4]=(_Float16)x1.x; h0[5]=(_Float16)x1.y; h0[6]=(_Float16)x1.z; h0[7]=(_Float16)x1.w;
      h1[0]=(_Float16)x2.x; h1[1]=(_Float16)x2.y; h1[2]=(_Float16)x2.z; h1[3]=(_Float16)x2.w;
      h1[4]=(_Float16)x3.x; h1[5]=(_Float16)x3.y; h1[6]=(_Float16)x3.z; h1[7]=(_Float16)x3.w;
      *(v8h*)&sX[xrow][xks]     = h0;
      *(v8h*)&sX[xrow][xks + 8] = h1;
    } else {
      *(v8h*)&sX[xrow][xks]     = ((const v8h*)xp)[0];
      *(v8h*)&sX[xrow][xks + 8] = ((const v8h*)xp)[1];
    }
    // ---- stage W tile (64 x 32 f16) ----
    *(v8h*)&sW[wrow][wks] =
        *(const v8h*)(W16 + (size_t)(bn0 + wrow) * K + k0 + wks);
    __syncthreads();

    v16h a0 = frag_a(&sX[wm * 32 +      l15][0], lane);
    v16h a1 = frag_a(&sX[wm * 32 + 16 + l15][0], lane);
    v16h b0 = frag_b(&sW[wn * 32 +      l15][0], lane);
    v16h b1 = frag_b(&sW[wn * 32 + 16 + l15][0], lane);
    acc[0][0] = __builtin_amdgcn_wmma_f32_16x16x32_f16(false, a0, false, b0, (short)0, acc[0][0], false, false);
    acc[0][1] = __builtin_amdgcn_wmma_f32_16x16x32_f16(false, a0, false, b1, (short)0, acc[0][1], false, false);
    acc[1][0] = __builtin_amdgcn_wmma_f32_16x16x32_f16(false, a1, false, b0, (short)0, acc[1][0], false, false);
    acc[1][1] = __builtin_amdgcn_wmma_f32_16x16x32_f16(false, a1, false, b1, (short)0, acc[1][1], false, false);
    __syncthreads();
  }

  // ---- epilogue: D-frag row = VGPR idx (+8 for upper half), col = lane&15 ----
  for (int i = 0; i < 2; ++i)
    for (int j = 0; j < 2; ++j) {
      const int rowb = bm0 + wm * 32 + i * 16 + half * 8;
      const int colg = bn0 + wn * 32 + j * 16 + l15;
      const float bv = bias[colg];
      if constexpr (TRANS_OUT) {
        v8h pack;
#pragma unroll
        for (int r = 0; r < 8; ++r)
          pack[r] = (_Float16)((acc[i][j][r] + bv) * scale);
        *(v8h*)((_Float16*)OutP + (size_t)colg * ldo + rowb) = pack;  // 16B store
      } else if constexpr (OUT_F32) {
        float* O = (float*)OutP;
#pragma unroll
        for (int r = 0; r < 8; ++r)
          O[(size_t)(rowb + r) * N + colg] = (acc[i][j][r] + bv) * scale;
      } else {
        _Float16* O = (_Float16*)OutP;
#pragma unroll
        for (int r = 0; r < 8; ++r)
          O[(size_t)(rowb + r) * N + colg] = (_Float16)((acc[i][j][r] + bv) * scale);
      }
    }
}

// ---------------------------------------------------------------------------
// Fused flash attention.  1 block per (b,h); 8 waves each own a 1024-key
// chunk with online softmax; partials merged through LDS.
// Qp already includes bias and 1/sqrt(HEAD_DIM).  Vpt is [HIDDEN][BLK] f16.
// ---------------------------------------------------------------------------
__global__ __launch_bounds__(256)
void attn_wmma(const _Float16* __restrict__ Qp, const _Float16* __restrict__ Kp,
               const _Float16* __restrict__ Vpt, _Float16* __restrict__ AttnOut) {
  const int bh   = blockIdx.x;
  const int b    = bh >> 4;
  const int h    = bh & 15;
  const int tid  = threadIdx.x;
  const int lane = tid & 31;
  const int w    = tid >> 5;
  const int l15  = lane & 15;
  const int half = lane >> 4;

  __shared__ alignas(16) _Float16 sP[8][16][40];   // per-wave P staging
  __shared__ float sO[8][16][64];                  // per-wave partial O
  __shared__ float sM[8][16];                      // per-wave row max
  __shared__ float sL[8][16];                      // per-wave row sum

  const _Float16* qrow = Qp + (size_t)(b * LQ + l15) * HIDDEN + h * HEAD_DIM;
  const v16h qa0 = frag_a(qrow, lane);        // head dims 0..31
  const v16h qa1 = frag_a(qrow + 32, lane);   // head dims 32..63

  v8f o0 = {}, o1 = {}, o2 = {}, o3 = {};
  float mrow[8], lrow[8];
#pragma unroll
  for (int r = 0; r < 8; ++r) { mrow[r] = -__builtin_inff(); lrow[r] = 0.f; }

  const int key0 = b * LK + w * (LK / 8);
  const size_t vstep = (size_t)16 * BLK;

  for (int j = 0; j < LK / 8; j += 32) {
    // ---- S = Q @ K^T, two 16(q) x 16(key) tiles over 32 keys ----
    const _Float16* kb0 = Kp + (size_t)(key0 + j +      l15) * HIDDEN + h * HEAD_DIM;
    const _Float16* kb1 = Kp + (size_t)(key0 + j + 16 + l15) * HIDDEN + h * HEAD_DIM;
    v16h bk00 = frag_b(kb0,      lane);
    v16h bk01 = frag_b(kb0 + 32, lane);
    v16h bk10 = frag_b(kb1,      lane);
    v16h bk11 = frag_b(kb1 + 32, lane);

    v8f s0 = {}, s1 = {};
    s0 = __builtin_amdgcn_wmma_f32_16x16x32_f16(false, qa0, false, bk00, (short)0, s0, false, false);
    s0 = __builtin_amdgcn_wmma_f32_16x16x32_f16(false, qa1, false, bk01, (short)0, s0, false, false);
    s1 = __builtin_amdgcn_wmma_f32_16x16x32_f16(false, qa0, false, bk10, (short)0, s1, false, false);
    s1 = __builtin_amdgcn_wmma_f32_16x16x32_f16(false, qa1, false, bk11, (short)0, s1, false, false);

    if (j + 32 < LK / 8) {   // pull next tiles toward L2 (global_prefetch_b8)
      __builtin_prefetch(Kp + (size_t)(key0 + j + 32 + l15) * HIDDEN + h * HEAD_DIM, 0, 0);
      __builtin_prefetch(Vpt + (size_t)(h * HEAD_DIM + l15) * (size_t)BLK + key0 + j + 32, 0, 0);
    }

    // ---- online softmax: row r lives in (VGPR r, lane half), 16 lanes/row ----
#pragma unroll
    for (int r = 0; r < 8; ++r) {
      float mx = fmaxf(s0[r], s1[r]);
      mx = fmaxf(mx, __shfl_xor(mx, 1));
      mx = fmaxf(mx, __shfl_xor(mx, 2));
      mx = fmaxf(mx, __shfl_xor(mx, 4));
      mx = fmaxf(mx, __shfl_xor(mx, 8));
      const float nm   = fmaxf(mrow[r], mx);
      const float corr = __expf(mrow[r] - nm);
      mrow[r] = nm;
      const float p0 = __expf(s0[r] - nm);
      const float p1 = __expf(s1[r] - nm);
      lrow[r] = lrow[r] * corr + p0 + p1;
      o0[r] *= corr; o1[r] *= corr; o2[r] *= corr; o3[r] *= corr;
      sP[w][r + half * 8][l15]      = (_Float16)p0;   // keys j..j+15
      sP[w][r + half * 8][16 + l15] = (_Float16)p1;   // keys j+16..j+31
    }

    // ---- O += P @ V (K = 32 keys, 4 dim tiles) ----
    const v16h pf = frag_a(&sP[w][l15][0], lane);     // in-wave DS order is safe
    const _Float16* vb = Vpt + (size_t)(h * HEAD_DIM + l15) * (size_t)BLK + key0 + j;
    v16h vf0 = frag_b(vb,             lane);
    v16h vf1 = frag_b(vb +     vstep, lane);
    v16h vf2 = frag_b(vb + 2 * vstep, lane);
    v16h vf3 = frag_b(vb + 3 * vstep, lane);
    o0 = __builtin_amdgcn_wmma_f32_16x16x32_f16(false, pf, false, vf0, (short)0, o0, false, false);
    o1 = __builtin_amdgcn_wmma_f32_16x16x32_f16(false, pf, false, vf1, (short)0, o1, false, false);
    o2 = __builtin_amdgcn_wmma_f32_16x16x32_f16(false, pf, false, vf2, (short)0, o2, false, false);
    o3 = __builtin_amdgcn_wmma_f32_16x16x32_f16(false, pf, false, vf3, (short)0, o3, false, false);
  }

  // ---- publish per-wave partials ----
#pragma unroll
  for (int r = 0; r < 8; ++r) {
    float l = lrow[r];
    l += __shfl_xor(l, 1); l += __shfl_xor(l, 2);
    l += __shfl_xor(l, 4); l += __shfl_xor(l, 8);
    lrow[r] = l;
    sO[w][r + half * 8][ 0 + l15] = o0[r];
    sO[w][r + half * 8][16 + l15] = o1[r];
    sO[w][r + half * 8][32 + l15] = o2[r];
    sO[w][r + half * 8][48 + l15] = o3[r];
  }
  if (l15 == 0) {
#pragma unroll
    for (int r = 0; r < 8; ++r) {
      sM[w][r + half * 8] = mrow[r];
      sL[w][r + half * 8] = lrow[r];
    }
  }
  __syncthreads();

  // ---- merge the 8 wave partials (max/sum-exp recombination) ----
  for (int i = tid; i < LQ * HEAD_DIM; i += 256) {
    const int row = i >> 6;
    const int dim = i & 63;
    float gm = -__builtin_inff();
#pragma unroll
    for (int ww = 0; ww < 8; ++ww) gm = fmaxf(gm, sM[ww][row]);
    float gl = 0.f, go = 0.f;
#pragma unroll
    for (int ww = 0; ww < 8; ++ww) {
      const float c = __expf(sM[ww][row] - gm);
      gl += sL[ww][row] * c;
      go += sO[ww][row][dim] * c;
    }
    AttnOut[(size_t)(b * LQ + row) * HIDDEN + h * HEAD_DIM + dim] =
        (_Float16)(go / gl);
  }
}

// ---------------------------------------------------------------------------
extern "C" void kernel_launch(void* const* d_in, const int* in_sizes, int n_in,
                              void* d_out, int out_size, void* d_ws,
                              size_t ws_size, hipStream_t stream) {
  (void)in_sizes; (void)n_in; (void)out_size; (void)ws_size;
  const float* query = (const float*)d_in[0];
  const float* key   = (const float*)d_in[1];
  const float* value = (const float*)d_in[2];
  const float* Wq = (const float*)d_in[3];
  const float* bq = (const float*)d_in[4];
  const float* Wk = (const float*)d_in[5];
  const float* bk = (const float*)d_in[6];
  const float* Wv = (const float*)d_in[7];
  const float* bv = (const float*)d_in[8];
  const float* Wo = (const float*)d_in[9];
  const float* bo = (const float*)d_in[10];

  char* ws = (char*)d_ws;
  size_t off = 0;
  auto take = [&](size_t bytes) {
    char* p = ws + off;
    off += (bytes + 255) & ~(size_t)255;
    return p;
  };
  _Float16* Wq16  = (_Float16*)take((size_t)HIDDEN * HIDDEN * 2);
  _Float16* Wk16  = (_Float16*)take((size_t)HIDDEN * HIDDEN * 2);
  _Float16* Wv16  = (_Float16*)take((size_t)HIDDEN * HIDDEN * 2);
  _Float16* Wo16  = (_Float16*)take((size_t)HIDDEN * HIDDEN * 2);
  _Float16* Qp    = (_Float16*)take((size_t)MQ * HIDDEN * 2);
  _Float16* AttnB = (_Float16*)take((size_t)MQ * HIDDEN * 2);
  _Float16* Kp    = (_Float16*)take((size_t)BLK * HIDDEN * 2);   // 128 MB
  _Float16* Vpt   = (_Float16*)take((size_t)HIDDEN * BLK * 2);   // 128 MB

  // 1) weights fp32 -> f16
  cvt_f32_f16<<<512, 256, 0, stream>>>(Wq, Wq16, HIDDEN * HIDDEN);
  cvt_f32_f16<<<512, 256, 0, stream>>>(Wk, Wk16, HIDDEN * HIDDEN);
  cvt_f32_f16<<<512, 256, 0, stream>>>(Wv, Wv16, HIDDEN * HIDDEN);
  cvt_f32_f16<<<512, 256, 0, stream>>>(Wo, Wo16, HIDDEN * HIDDEN);

  // 2) projections (Q carries the 1/sqrt(HEAD_DIM) softmax scale)
  gemm_wmma<float, false, false><<<dim3(MQ / 128, HIDDEN / 64), 256, 0, stream>>>(
      query, Wq16, bq, Qp, MQ, HIDDEN, HIDDEN, 0, 0.125f);
  gemm_wmma<float, false, false><<<dim3(BLK / 128, HIDDEN / 64), 256, 0, stream>>>(
      key, Wk16, bk, Kp, BLK, HIDDEN, HIDDEN, 0, 1.0f);
  gemm_wmma<float, false, true><<<dim3(BLK / 128, HIDDEN / 64), 256, 0, stream>>>(
      value, Wv16, bv, Vpt, BLK, HIDDEN, HIDDEN, BLK, 1.0f);

  // 3) fused flash attention: one block per (batch, head)
  attn_wmma<<<BATCH * HEADS, 256, 0, stream>>>(Qp, Kp, Vpt, AttnB);

  // 4) output projection, f32 result
  gemm_wmma<_Float16, true, false><<<dim3(MQ / 128, HIDDEN / 64), 256, 0, stream>>>(
      AttnB, Wo16, bo, d_out, MQ, HIDDEN, HIDDEN, 0, 1.0f);
}
